// SudokuPasses_17145509446028
// MI455X (gfx1250) — compile-verified, matrix operand order checked by reading, not run on hardware
//
#include <hip/hip_runtime.h>
#include <cstdint>

typedef __attribute__((ext_vector_type(16))) _Float16 v16h;
typedef __attribute__((ext_vector_type(8)))  float    v8f;

#define W27  0x07FFFFFFu   // 27 valid bits (3 rows x 9 cols)
#define F9   0x000001FFu   // one 9-bit row field
#define RSEL 0x00040201u   // bits 0,9,18 (one bit per row in a word)
#define G9   0x00000049u   // bits 0,3,6 (one bit per 3-bit group in a 9-bit field)
#define C3   0x01249249u   // every 3rd bit over 27 bits

// ---------- small bit helpers -------------------------------------------------

static __device__ __forceinline__ uint32_t grp_any9(uint32_t x){ return (x|(x>>1)|(x>>2)) & G9; }
static __device__ __forceinline__ uint32_t grp_ge29(uint32_t x){ return ((x&(x>>1))|(x&(x>>2))|((x>>1)&(x>>2))) & G9; }
static __device__ __forceinline__ uint32_t grp_par9(uint32_t x){ return (x^(x>>1)^(x>>2)) & G9; }
static __device__ __forceinline__ uint32_t bc3(uint32_t x){ return x * 7u; }                 // bit 3k -> bits 3k..3k+2
static __device__ __forceinline__ uint32_t rows3(uint32_t x){ return x | (x<<9) | (x<<18); } // row0 9b -> all 3 rows

// per-cell candidate counts across the 9 digit planes (carry-save popcount)
static __device__ __forceinline__ void cell_counts(const uint32_t m[9][3],
                                                   uint32_t ex1[3], uint32_t ex2[3]) {
  #pragma unroll
  for (int w = 0; w < 3; ++w) {
    uint32_t ones = 0, twos = 0, four = 0;
    #pragma unroll
    for (int d = 0; d < 9; ++d) {
      uint32_t p  = m[d][w];
      uint32_t c1 = ones & p;  ones ^= p;
      uint32_t c2 = twos & c1; twos ^= c1;
      four |= c2;
    }
    ex1[w] = ones & ~twos & ~four;   // count == 1
    ex2[w] = twos & ~ones & ~four;   // count == 2
  }
}

// per-box (3x3) statistics of a word; outputs broadcast to the full 9 cells of each box
static __device__ __forceinline__ void box_stats(uint32_t w, uint32_t* anyB, uint32_t* ex1B) {
  uint32_t u0 = w & F9, u1 = (w >> 9) & F9, u2 = (w >> 18) & F9;
  uint32_t cge2 = (u0&u1)|(u0&u2)|(u1&u2);     // column-of-box count >= 2
  uint32_t h    = u0 | u1 | u2;                // column-of-box count >= 1
  uint32_t aG   = grp_any9(cge2);
  uint32_t anyb = grp_any9(h);
  uint32_t ex1b = grp_par9(h) & ~grp_ge29(h) & ~aG;   // total count in box == 1
  *anyB = rows3(bc3(anyb));
  *ex1B = rows3(bc3(ex1b));
}

// per-box "count >= 2" broadcast mask
static __device__ __forceinline__ uint32_t box_ge2B(uint32_t w) {
  uint32_t u0 = w & F9, u1 = (w >> 9) & F9, u2 = (w >> 18) & F9;
  uint32_t cge2 = (u0&u1)|(u0&u2)|(u1&u2);
  uint32_t h    = u0 | u1 | u2;
  uint32_t b = grp_any9(cge2) | grp_ge29(h);
  return rows3(bc3(b));
}

// per-row "exactly one bit set" broadcast mask (bit-sliced popcount over 9 columns)
static __device__ __forceinline__ uint32_t row_ex1_bcast(uint32_t w) {
  uint32_t ones = 0, twos = 0;
  #pragma unroll
  for (int c = 0; c < 9; ++c) { uint32_t p = (w >> c) & RSEL; twos |= ones & p; ones ^= p; }
  return (ones & ~twos) * F9;      // bits 0,9,18 -> whole rows
}

// per-column (9 rows across 3 words) exactly-one / ge2, 9-bit results
static __device__ __forceinline__ uint32_t col_ex1(const uint32_t w[3]) {
  uint32_t ones = 0, twos = 0;
  #pragma unroll
  for (int i = 0; i < 3; ++i) {
    uint32_t a = w[i] & F9, b = (w[i] >> 9) & F9, c = (w[i] >> 18) & F9;
    twos |= ones & a; ones ^= a;
    twos |= ones & b; ones ^= b;
    twos |= ones & c; ones ^= c;
  }
  return ones & ~twos;
}
static __device__ __forceinline__ uint32_t col_ge2(const uint32_t w[3]) {
  uint32_t ones = 0, twos = 0;
  #pragma unroll
  for (int i = 0; i < 3; ++i) {
    uint32_t a = w[i] & F9, b = (w[i] >> 9) & F9, c = (w[i] >> 18) & F9;
    twos |= ones & a; ones ^= a;
    twos |= ones & b; ones ^= b;
    twos |= ones & c; ones ^= c;
  }
  return twos;
}

// ---------- the nine propagation steps ---------------------------------------

static __device__ __forceinline__ void step_filter_box(uint32_t m[9][3]) {
  uint32_t ex1[3], ex2[3];
  cell_counts(m, ex1, ex2);
  #pragma unroll
  for (int d = 0; d < 9; ++d)
    #pragma unroll
    for (int w = 0; w < 3; ++w) {
      uint32_t s = m[d][w] & ex1[w];      // solved cells carrying digit d
      uint32_t anyB, e1B;
      box_stats(s, &anyB, &e1B);
      // keep iff: no solved d in box, or exactly one and it's this cell
      m[d][w] &= (~anyB) | (e1B & s);
    }
}

static __device__ __forceinline__ void step_pointing_h(uint32_t m[9][3]) {
  #pragma unroll
  for (int d = 0; d < 9; ++d)
    #pragma unroll
    for (int w = 0; w < 3; ++w) {
      uint32_t v = m[d][w];
      uint32_t pres = (v | (v >> 1) | (v >> 2)) & C3;   // presence(row r, box bc) at bit 9r+3bc
      uint32_t x0 = pres & G9, x1 = (pres >> 9) & G9, x2 = (pres >> 18) & G9;
      uint32_t g2  = (x0&x1)|(x0&x2)|(x1&x2);
      uint32_t ex1 = (x0^x1^x2) & ~g2;                  // digit confined to 1 row within box
      uint32_t p[3] = { x0 & ex1, x1 & ex1, x2 & ex1 };
      uint32_t rem = 0;
      #pragma unroll
      for (int r = 0; r < 3; ++r) {
        uint32_t s = p[r];                              // pointing boxes for this band-row
        uint32_t a  = (s | (s>>3) | (s>>6)) & 1u;
        uint32_t gg = ((s&(s>>3)) | (s&(s>>6)) | ((s>>3)&(s>>6))) & 1u;
        uint32_t e1 = ((s ^ (s>>3) ^ (s>>6)) & 1u) & ~gg;
        uint32_t remb = (a * G9) & ~((e1 * G9) & s);    // boxes losing digit d in this row
        rem |= (bc3(remb)) << (9 * r);
      }
      m[d][w] = v & ~rem;
    }
}

static __device__ __forceinline__ void step_pointing_v(uint32_t m[9][3]) {
  #pragma unroll
  for (int d = 0; d < 9; ++d) {
    uint32_t s[3];
    #pragma unroll
    for (int w = 0; w < 3; ++w) {
      uint32_t x  = m[d][w];
      uint32_t cp = (x | (x >> 9) | (x >> 18)) & F9;        // presence(box bc, col ci)
      uint32_t e  = grp_par9(cp) & ~grp_ge29(cp);           // exactly one col inside box
      s[w] = cp & bc3(e);                                   // pointing column bits
    }
    uint32_t any = s[0] | s[1] | s[2];
    uint32_t g2  = (s[0]&s[1]) | (s[0]&s[2]) | (s[1]&s[2]);
    uint32_t e1  = (s[0]^s[1]^s[2]) & ~g2;
    #pragma unroll
    for (int w = 0; w < 3; ++w) {
      uint32_t remb = any & ~(e1 & s[w]);
      m[d][w] &= ~(remb * RSEL);
    }
  }
}

static __device__ __forceinline__ void step_unique_h(uint32_t m[9][3]) {
  uint32_t hid[9][3], has[3] = {0,0,0};
  #pragma unroll
  for (int d = 0; d < 9; ++d)
    #pragma unroll
    for (int w = 0; w < 3; ++w) {
      uint32_t h = m[d][w] & row_ex1_bcast(m[d][w]);
      hid[d][w] = h; has[w] |= h;
    }
  #pragma unroll
  for (int d = 0; d < 9; ++d)
    #pragma unroll
    for (int w = 0; w < 3; ++w)
      m[d][w] = hid[d][w] | (m[d][w] & ~has[w]);
}

static __device__ __forceinline__ void step_unique_v(uint32_t m[9][3]) {
  uint32_t hid[9][3], has[3] = {0,0,0};
  #pragma unroll
  for (int d = 0; d < 9; ++d) {
    uint32_t colB = col_ex1(m[d]) * RSEL;
    #pragma unroll
    for (int w = 0; w < 3; ++w) {
      uint32_t h = m[d][w] & colB;
      hid[d][w] = h; has[w] |= h;
    }
  }
  #pragma unroll
  for (int d = 0; d < 9; ++d)
    #pragma unroll
    for (int w = 0; w < 3; ++w)
      m[d][w] = hid[d][w] | (m[d][w] & ~has[w]);
}

static __device__ __forceinline__ void step_unique_box(uint32_t m[9][3]) {
  uint32_t hid[9][3], has[3] = {0,0,0};
  #pragma unroll
  for (int d = 0; d < 9; ++d)
    #pragma unroll
    for (int w = 0; w < 3; ++w) {
      uint32_t anyB, e1B;
      box_stats(m[d][w], &anyB, &e1B);
      uint32_t h = m[d][w] & e1B;
      hid[d][w] = h; has[w] |= h;
    }
  #pragma unroll
  for (int d = 0; d < 9; ++d)
    #pragma unroll
    for (int w = 0; w < 3; ++w)
      m[d][w] = hid[d][w] | (m[d][w] & ~has[w]);
}

static __device__ __forceinline__ void step_doubles_v(uint32_t m[9][3]) {
  uint32_t ex1[3], ex2[3];
  cell_counts(m, ex1, ex2);
  uint32_t pm[9][3];
  #pragma unroll
  for (int d = 0; d < 9; ++d)
    #pragma unroll
    for (int w = 0; w < 3; ++w) pm[d][w] = m[d][w] & ex2[w];

  uint32_t fd[9] = {0,0,0,0,0,0,0,0,0};   // per digit: columns where digit is in a naked pair
  uint32_t FP[3] = {0,0,0};               // flagged pair cells
  #pragma unroll
  for (int i = 0; i < 9; ++i)
    #pragma unroll
    for (int j = i + 1; j < 9; ++j) {
      uint32_t P[3] = { pm[i][0] & pm[j][0], pm[i][1] & pm[j][1], pm[i][2] & pm[j][2] };
      uint32_t flag = col_ge2(P);         // columns holding >=2 identical {i,j} cells
      fd[i] |= flag; fd[j] |= flag;
      uint32_t cB = flag * RSEL;
      #pragma unroll
      for (int w = 0; w < 3; ++w) FP[w] |= cB & P[w];
    }
  #pragma unroll
  for (int d = 0; d < 9; ++d) {
    uint32_t cBd = fd[d] * RSEL;
    #pragma unroll
    for (int w = 0; w < 3; ++w)
      m[d][w] &= ~(cBd & ~(pm[d][w] & FP[w]));
  }
}

static __device__ __forceinline__ void step_doubles_box(uint32_t m[9][3]) {
  uint32_t ex1[3], ex2[3];
  cell_counts(m, ex1, ex2);
  uint32_t pm[9][3];
  #pragma unroll
  for (int d = 0; d < 9; ++d)
    #pragma unroll
    for (int w = 0; w < 3; ++w) pm[d][w] = m[d][w] & ex2[w];

  uint32_t fdB[9][3];
  #pragma unroll
  for (int d = 0; d < 9; ++d) { fdB[d][0]=0; fdB[d][1]=0; fdB[d][2]=0; }
  uint32_t FP[3] = {0,0,0};
  #pragma unroll
  for (int i = 0; i < 9; ++i)
    #pragma unroll
    for (int j = i + 1; j < 9; ++j) {
      #pragma unroll
      for (int w = 0; w < 3; ++w) {
        uint32_t P  = pm[i][w] & pm[j][w];
        uint32_t g2 = box_ge2B(P);
        fdB[i][w] |= g2; fdB[j][w] |= g2;
        FP[w] |= g2 & P;
      }
    }
  #pragma unroll
  for (int d = 0; d < 9; ++d)
    #pragma unroll
    for (int w = 0; w < 3; ++w)
      m[d][w] &= ~(fdB[d][w] & ~(pm[d][w] & FP[w]));
}

// ---------- kernels -----------------------------------------------------------

__global__ void sudoku_pack(const float* __restrict__ in, uint32_t* __restrict__ out, int n) {
  int t = blockIdx.x * blockDim.x + threadIdx.x;   // t = b*9 + d
  if (t >= n) return;
  const float* src = in + (size_t)t * 81;
  uint32_t w0 = 0, w1 = 0, w2 = 0;
  #pragma unroll
  for (int i = 0; i < 81; ++i) {
    int r = i / 9, c = i % 9;
    int wi = r / 3, bit = (r - wi * 3) * 9 + c;
    uint32_t b = (src[i] != 0.0f) ? (1u << bit) : 0u;
    if      (wi == 0) w0 |= b;
    else if (wi == 1) w1 |= b;
    else              w2 |= b;
  }
  out[t * 3 + 0] = w0; out[t * 3 + 1] = w1; out[t * 3 + 2] = w2;
}

__global__ void sudoku_unpack(const uint32_t* __restrict__ in, float* __restrict__ out, int n) {
  int t = blockIdx.x * blockDim.x + threadIdx.x;
  if (t >= n) return;
  uint32_t w0 = in[t * 3 + 0], w1 = in[t * 3 + 1], w2 = in[t * 3 + 2];
  float* dst = out + (size_t)t * 81;
  #pragma unroll
  for (int i = 0; i < 81; ++i) {
    int r = i / 9, c = i % 9;
    int wi = r / 3, bit = (r - wi * 3) * 9 + c;
    uint32_t w = (wi == 0) ? w0 : ((wi == 1) ? w1 : w2);
    dst[i] = ((w >> bit) & 1u) ? 1.0f : 0.0f;
  }
}

static __device__ __forceinline__ v16h bits_to_v16h(uint32_t bits) {
  v16h a;
  #pragma unroll
  for (int j = 0; j < 16; ++j)
    a[j] = ((bits >> j) & 1u) ? (_Float16)1.0f : (_Float16)0.0f;
  return a;
}

__global__ void sudoku_solve(const uint32_t* __restrict__ pin, uint32_t* __restrict__ pout,
                             float* __restrict__ solvedOut, int B) {
  int b = blockIdx.x * blockDim.x + threadIdx.x;
  if (b >= B) b = B - 1;               // keep waves fully populated for WMMA (benign duplicate)

  uint32_t m[9][3];
  #pragma unroll
  for (int d = 0; d < 9; ++d)
    #pragma unroll
    for (int w = 0; w < 3; ++w)
      m[d][w] = pin[((size_t)b * 9 + d) * 3 + w] & W27;

  #pragma unroll 1
  for (int pass = 0; pass < 4; ++pass) {
    step_filter_box(m);
    step_pointing_h(m);
    step_pointing_v(m);
    step_unique_h(m);
    step_unique_v(m);
    step_unique_box(m);
    #pragma unroll 1
    for (int rep = 0; rep < 2; ++rep) step_doubles_v(m);
    step_doubles_box(m);
  }

  #pragma unroll
  for (int d = 0; d < 9; ++d)
    #pragma unroll
    for (int w = 0; w < 3; ++w)
      pout[((size_t)b * 9 + d) * 3 + w] = m[d][w];

  // ---- solved flag via v_wmma_f32_16x16x32_f16: popcount(exactly-one mask) == 81 ----
  uint32_t ex1[3], ex2u[3];
  cell_counts(m, ex1, ex2u);

  int lane = (int)(threadIdx.x & 31u);
  v16h onesv;
  #pragma unroll
  for (int j = 0; j < 16; ++j) onesv[j] = (_Float16)1.0f;

  v8f acc0 = {}, acc1 = {};
  #pragma unroll
  for (int t = 0; t < 3; ++t) {
    uint32_t wrd = ex1[t];
    // set 0: rows m = puzzles of lanes 0..15 (lane m: low 16 bits, lane m+16: high bits)
    uint32_t v0 = (uint32_t)__shfl((int)wrd, lane & 15, 32);
    uint32_t bits0 = (lane < 16) ? (v0 & 0xFFFFu) : (v0 >> 16);
    // set 1: rows m = puzzles of lanes 16..31
    uint32_t v1 = (uint32_t)__shfl((int)wrd, 16 | (lane & 15), 32);
    uint32_t bits1 = (lane < 16) ? (v1 & 0xFFFFu) : (v1 >> 16);

    v16h a0 = bits_to_v16h(bits0);
    v16h a1 = bits_to_v16h(bits1);
    acc0 = __builtin_amdgcn_wmma_f32_16x16x32_f16(false, a0, false, onesv, (short)0, acc0, false, false);
    acc1 = __builtin_amdgcn_wmma_f32_16x16x32_f16(false, a1, false, onesv, (short)0, acc1, false, false);
  }

  // D[m,n] is constant over n; lane q holds D[(q&7)+(q>=16?8:0), q&15] in acc[q&7]
  float s0 = 0.0f, s1 = 0.0f;
  #pragma unroll
  for (int j = 0; j < 8; ++j) {
    s0 += ((lane & 7) == j) ? acc0[j] : 0.0f;
    s1 += ((lane & 7) == j) ? acc1[j] : 0.0f;
  }
  int srcl = (lane & 7) | ((lane & 8) << 1);
  float t0 = __shfl(s0, srcl, 32);
  float t1 = __shfl(s1, srcl, 32);
  float mysum = (lane < 16) ? t0 : t1;

  solvedOut[b] = (mysum == 81.0f) ? 1.0f : 0.0f;
}

// ---------- host entry --------------------------------------------------------

extern "C" void kernel_launch(void* const* d_in, const int* in_sizes, int n_in,
                              void* d_out, int out_size, void* d_ws, size_t ws_size,
                              hipStream_t stream) {
  (void)n_in; (void)out_size; (void)ws_size;
  const float* mask = (const float*)d_in[0];
  float* out = (float*)d_out;
  int B = in_sizes[0] / 729;

  uint32_t* packedIn  = (uint32_t*)d_ws;
  uint32_t* packedOut = packedIn + (size_t)B * 27;

  int nPD = B * 9;
  sudoku_pack<<<(nPD + 255) / 256, 256, 0, stream>>>(mask, packedIn, nPD);
  sudoku_solve<<<(B + 255) / 256, 256, 0, stream>>>(packedIn, packedOut,
                                                    out + (size_t)B * 729, B);
  sudoku_unpack<<<(nPD + 255) / 256, 256, 0, stream>>>(packedOut, out, nPD);
}
